// WindowMHSA1D_18571438588115
// MI455X (gfx1250) — compile-verified, matrix operand order checked by reading, not run on hardware
//
#include <hip/hip_runtime.h>

#define T_SEQ   1024
#define C_DIM   256
#define N_HEADS 4
#define HDIM    64
#define WIN_W   128
#define PAD_W   63
#define SHIFT_W 64
#define L_WIN   1023          /* T + 2*PAD - WIN + 1 */
#define TPA     1152          /* padded time alloc (>= 1150) */
#define E3      768

typedef unsigned short u16t;
typedef unsigned int   u32t;
typedef unsigned long long u64t;
typedef __attribute__((ext_vector_type(16))) __bf16 v16bf;
typedef __attribute__((ext_vector_type(8)))  float  v8f;

/* ---------------- bf16 helpers (RNE) ---------------- */
static __device__ __forceinline__ u16t f2bfu(float f){
    u32t u = __builtin_bit_cast(u32t, f);
    u += 0x7fffu + ((u >> 16) & 1u);
    return (u16t)(u >> 16);
}
static __device__ __forceinline__ float bfu2f(u16t h){
    u32t u = ((u32t)h) << 16;
    return __builtin_bit_cast(float, u);
}
static __device__ __forceinline__ __bf16 u2bf(u16t h){
    return __builtin_bit_cast(__bf16, h);
}
static __device__ __forceinline__ v8f vzero8(){
    v8f z;
#pragma unroll
    for (int i = 0; i < 8; ++i) z[i] = 0.0f;
    return z;
}
static __device__ __forceinline__ v16bf frag_zero(){
    v16bf f;
#pragma unroll
    for (int i = 0; i < 16; ++i) f[i] = u2bf(0);
    return f;
}

/* Fragment loader for v_wmma_f32_16x16x32_bf16 operands.
 * ISA 16-bit A-layout (and symmetric B as N x K row-major):
 *   lane half kh = lane>>4; element e: K = (e/8)*16 + kh*8 + (e%8).
 * Caller passes the per-lane row (A: M = lane&15; B: N = lane&15). */
static __device__ __forceinline__ v16bf frag_ld(const u16t* __restrict__ base, int stride,
                                                int row, int kbase, int lane){
    const u16t* rp = base + (size_t)row * stride + kbase + ((lane >> 4) << 3);
    v16bf f;
#pragma unroll
    for (int g = 0; g < 2; ++g)
#pragma unroll
        for (int e = 0; e < 8; ++e)
            f[g * 8 + e] = u2bf(rp[g * 16 + e]);
    return f;
}
static __device__ __forceinline__ v8f wmma_bf16(v16bf a, v16bf b, v8f c){
    return __builtin_amdgcn_wmma_f32_16x16x32_bf16(false, a, false, b, (short)0, c, false, false);
}
static __device__ __forceinline__ float hann_w(int n){
    return 0.5f * (1.0f - __cosf(6.2831853071795864f * (float)n / 127.0f));
}
/* number of windows covering padded index p */
static __device__ __forceinline__ float win_count(int p){
    int lo = p - (WIN_W - 1); if (lo < 0) lo = 0;
    int hi = p; if (hi > L_WIN - 1) hi = L_WIN - 1;
    return (float)(hi - lo + 1);
}

/* ---------------- utility kernels ---------------- */
__global__ void k_zero(float* __restrict__ p, int n){
    for (int i = blockIdx.x * blockDim.x + threadIdx.x; i < n; i += gridDim.x * blockDim.x)
        p[i] = 0.0f;
}
__global__ void k_cvt_f2b(const float* __restrict__ s, u16t* __restrict__ d, int n){
    for (int i = blockIdx.x * blockDim.x + threadIdx.x; i < n; i += gridDim.x * blockDim.x)
        d[i] = f2bfu(s[i]);
}

/* ---------------- LayerNorm (per time step) -> bf16 ---------------- */
__global__ __launch_bounds__(256) void k_layernorm(const float* __restrict__ x,
        const float* __restrict__ g, const float* __restrict__ b,
        u16t* __restrict__ ylnb){
    int t = blockIdx.x, c = threadIdx.x;
    float v = x[t * C_DIM + c];
    __shared__ float sh[256];
    sh[c] = v; __syncthreads();
#pragma unroll
    for (int o = 128; o > 0; o >>= 1){ if (c < o) sh[c] += sh[c + o]; __syncthreads(); }
    float mu = sh[0] * (1.0f / 256.0f); __syncthreads();
    float d = v - mu;
    sh[c] = d * d; __syncthreads();
#pragma unroll
    for (int o = 128; o > 0; o >>= 1){ if (c < o) sh[c] += sh[c + o]; __syncthreads(); }
    float var = sh[0] * (1.0f / 256.0f);
    ylnb[t * C_DIM + c] = f2bfu(d * rsqrtf(var + 1e-5f) * g[c] + b[c]);
}

/* ---------------- QKV projection over padded/rolled time ----------------
 * QKp[v][p][0:512]  = Q|K channels (bf16, zeros in the pad rows)
 * VpT[v][c][p]      = V transposed (bf16) for row-contiguous B fragments  */
__global__ __launch_bounds__(256) void k_qkv_proj(const u16t* __restrict__ ylnb,
        const u16t* __restrict__ Wqkv, u16t* __restrict__ QKp, u16t* __restrict__ VpT){
    int lane = threadIdx.x & 31, wv = threadIdx.x >> 5;
    int tile = blockIdx.x * 8 + wv;                 /* 2 * 72 * 48 = 6912 tiles */
    int var = tile / (72 * 48);
    int rem = tile % (72 * 48);
    int pt = rem / 48, et = rem % 48;

    int pA = pt * 16 + (lane & 15);
    int tA = pA - PAD_W;
    bool valid = (tA >= 0) && (tA < T_SEQ);
    int src = 0;
    if (valid) src = var ? ((tA - SHIFT_W + T_SEQ) & (T_SEQ - 1)) : tA;

    v8f acc = vzero8();
#pragma unroll
    for (int ks = 0; ks < 8; ++ks){
        v16bf a;
        if (valid) a = frag_ld(ylnb, C_DIM, src, ks * 32, lane);
        else       a = frag_zero();
        v16bf bw = frag_ld(Wqkv, C_DIM, et * 16 + (lane & 15), ks * 32, lane);
        acc = wmma_bf16(a, bw, acc);
    }
    int half = lane >> 4;
#pragma unroll
    for (int vv = 0; vv < 8; ++vv){
        int p = pt * 16 + vv + half * 8;
        int e = et * 16 + (lane & 15);
        u16t hv = f2bfu(acc[vv]);
        if (e < 512) QKp[((size_t)var * TPA + p) * 512 + e] = hv;
        else         VpT[((size_t)var * C_DIM + (e - 512)) * TPA + p] = hv;
    }
}

/* ---------------- bias rank-1 score corrections ----------------
 * u[v][p][h] = Qp[p]·bk_h ; w[v][p][h] = bq_h·Kp[p] */
__global__ __launch_bounds__(256) void k_biasdots(const u16t* __restrict__ QKp,
        const float* __restrict__ inb, float* __restrict__ ubuf, float* __restrict__ wbuf){
    int var = blockIdx.x / TPA, p = blockIdx.x % TPA;
    int h = threadIdx.x >> 6, c = threadIdx.x & 63;
    __shared__ float sh[256];
    const u16t* row = QKp + ((size_t)var * TPA + p) * 512;
    float qv = bfu2f(row[h * 64 + c]);
    float kv = bfu2f(row[256 + h * 64 + c]);
    float bq = inb[h * 64 + c], bk = inb[256 + h * 64 + c];
    sh[threadIdx.x] = qv * bk; __syncthreads();
#pragma unroll
    for (int o = 32; o > 0; o >>= 1){ if (c < o) sh[threadIdx.x] += sh[threadIdx.x + o]; __syncthreads(); }
    if (c == 0) ubuf[((size_t)var * TPA + p) * N_HEADS + h] = sh[h * 64];
    __syncthreads();
    sh[threadIdx.x] = kv * bq; __syncthreads();
#pragma unroll
    for (int o = 32; o > 0; o >>= 1){ if (c < o) sh[threadIdx.x] += sh[threadIdx.x + o]; __syncthreads(); }
    if (c == 0) wbuf[((size_t)var * TPA + p) * N_HEADS + h] = sh[h * 64];
}
__global__ __launch_bounds__(256) void k_c0(const float* __restrict__ inb, float* __restrict__ c0){
    int h = threadIdx.x >> 6, c = threadIdx.x & 63;
    __shared__ float sh[256];
    sh[threadIdx.x] = inb[h * 64 + c] * inb[256 + h * 64 + c]; __syncthreads();
#pragma unroll
    for (int o = 32; o > 0; o >>= 1){ if (c < o) sh[threadIdx.x] += sh[threadIdx.x + o]; __syncthreads(); }
    if (c == 0) c0[h] = sh[h * 64];
}

/* ---------------- windowed attention core ----------------
 * one workgroup = (variant, window l, head h); 8 waves, wave i owns rows 16i..16i+15.
 * K strip is staged into LDS with CDNA5 async LDS loads (ASYNCcnt),
 * bypassing VGPRs and overlapping with the V strip prefetch.              */
__global__ __launch_bounds__(256) void k_attn(const u16t* __restrict__ QKp,
        const u16t* __restrict__ VpT, const float* __restrict__ ubuf,
        const float* __restrict__ wbuf, const float* __restrict__ c0,
        const float* __restrict__ inb, float* __restrict__ acc){
    int bid = blockIdx.x;
    int var = bid / (L_WIN * N_HEADS);
    int rem = bid % (L_WIN * N_HEADS);
    int l = rem / N_HEADS, h = rem % N_HEADS;
    int tid = threadIdx.x, lane = tid & 31, iq = tid >> 5, half = lane >> 4;

    __shared__ __align__(16) u16t Ks[WIN_W][72];   /* key strip, 144B rows */
    __shared__ __align__(16) u16t Ps[8][16][136];  /* per-wave softmax strip */

    const u16t* QK = QKp + (size_t)var * TPA * 512;

    /* async memory->LDS staging of the 128x64 bf16 key strip.
     * segment = 16 bytes; global addr: base + ((l+j)*512 + 256 + h*64 + s*8)*2
     * is 16B aligned (row stride 1024B, lane offsets 16B).               */
    {
        unsigned ksbase = (unsigned)(size_t)(&Ks[0][0]);
        u64t     gbase  = (u64t)(const void*)QK;
        for (int idx = tid; idx < WIN_W * 8; idx += 256){
            int j = idx >> 3, s = idx & 7;
            unsigned lds  = ksbase + (unsigned)j * 144u + (unsigned)s * 16u;
            unsigned voff = (unsigned)((((l + j) << 9) + 256 + (h << 6) + (s << 3)) << 1);
            asm volatile("global_load_async_to_lds_b128 %0, %1, %2"
                         :: "v"(lds), "v"(voff), "s"(gbase) : "memory");
        }
    }

    /* prefetch the V strip lines (2B-aligned rows -> scalar gathers later) */
    {
        const u16t* vpre = VpT + ((size_t)var * C_DIM + h * 64) * TPA + l;
        for (int r = tid >> 2; r < HDIM; r += 64)
            __builtin_prefetch(vpre + (size_t)r * TPA + ((tid & 3) << 5), 0, 3);
    }

    asm volatile("s_wait_asynccnt 0" ::: "memory");
    __syncthreads();

    /* S = Qp Kp^T (Gram), 16x128 strip per wave in registers */
    v8f S[8];
#pragma unroll
    for (int jt = 0; jt < 8; ++jt) S[jt] = vzero8();
    const u16t* Qbase = QK + (size_t)(l + iq * 16) * 512 + h * 64;
#pragma unroll
    for (int ks = 0; ks < 2; ++ks){
        v16bf a = frag_ld(Qbase, 512, lane & 15, ks * 32, lane);
#pragma unroll
        for (int jt = 0; jt < 8; ++jt){
            v16bf bk = frag_ld(&Ks[0][0], 72, jt * 16 + (lane & 15), ks * 32, lane);
            S[jt] = wmma_bf16(a, bk, S[jt]);
        }
    }

    /* epilogue: hann + bias corrections + scale */
    const float scale = 0.125f;           /* 1/sqrt(64) */
    float hj[8], wj[8];
#pragma unroll
    for (int jt = 0; jt < 8; ++jt){
        int j = jt * 16 + (lane & 15);
        hj[jt] = hann_w(j);
        wj[jt] = wbuf[((size_t)var * TPA + l + j) * N_HEADS + h];
    }
    float cc0 = c0[h];
#pragma unroll
    for (int vv = 0; vv < 8; ++vv){
        int i = iq * 16 + vv + half * 8;
        float hi = hann_w(i);
        float ui = ubuf[((size_t)var * TPA + l + i) * N_HEADS + h];
#pragma unroll
        for (int jt = 0; jt < 8; ++jt)
            S[jt][vv] = scale * (hi * hj[jt] * S[jt][vv] + hi * ui + hj[jt] * wj[jt] + cc0);
    }

    /* softmax per row: reduce across the 16-lane half + 8 column tiles   */
#pragma unroll
    for (int vv = 0; vv < 8; ++vv){
        float m = -3.4e38f;
#pragma unroll
        for (int jt = 0; jt < 8; ++jt) m = fmaxf(m, S[jt][vv]);
#pragma unroll
        for (int o = 1; o < 16; o <<= 1) m = fmaxf(m, __shfl_xor(m, o, 32));
        float sum = 0.0f;
#pragma unroll
        for (int jt = 0; jt < 8; ++jt){ float e = __expf(S[jt][vv] - m); S[jt][vv] = e; sum += e; }
#pragma unroll
        for (int o = 1; o < 16; o <<= 1) sum += __shfl_xor(sum, o, 32);
        float inv = 1.0f / sum;
        int i_loc = vv + half * 8;
#pragma unroll
        for (int jt = 0; jt < 8; ++jt){
            float p = S[jt][vv] * inv * hj[jt];          /* fold hann_j into P */
            Ps[iq][i_loc][jt * 16 + (lane & 15)] = f2bfu(p);
        }
    }
    __syncthreads();

    /* O = P' @ Vp  (16 x 64 per wave), accumulate overlapping windows    */
    const u16t* Vb = VpT + ((size_t)var * C_DIM + h * 64) * TPA + l;
#pragma unroll
    for (int ct = 0; ct < 4; ++ct){
        v8f O = vzero8();
#pragma unroll
        for (int ks = 0; ks < 4; ++ks){
            v16bf a = frag_ld(&Ps[iq][0][0], 136, lane & 15, ks * 32, lane);
            v16bf bv = frag_ld(Vb, TPA, ct * 16 + (lane & 15), ks * 32, lane);
            O = wmma_bf16(a, bv, O);
        }
        int cch = ct * 16 + (lane & 15);
        float bvv = inb[512 + h * 64 + cch];             /* sum_j P = 1 */
#pragma unroll
        for (int vv = 0; vv < 8; ++vv){
            int p = l + iq * 16 + vv + half * 8;
            atomicAdd(&acc[((size_t)var * TPA + p) * C_DIM + h * 64 + cch], O[vv] + bvv);
        }
    }
}

/* ---------------- out-projection + count normalize + combine variants ---------------- */
__global__ __launch_bounds__(128) void k_outproj(const u16t* __restrict__ accb,
        const u16t* __restrict__ Woutb, const float* __restrict__ outb,
        float* __restrict__ ycomb, u16t* __restrict__ ycombb){
    int lane = threadIdx.x & 31, wv = threadIdx.x >> 5, half = lane >> 4;
    int tile = blockIdx.x * 4 + wv;                      /* 64 x 16 tiles */
    int tt = tile >> 4, cc = tile & 15;
    int mA = lane & 15;
    int t0 = tt * 16 + mA;
    int p0 = PAD_W + t0;
    int p1 = PAD_W + ((t0 + SHIFT_W) & (T_SEQ - 1));     /* roll(y_shift, -SHIFT) */
    v8f a0 = vzero8(), a1 = vzero8();
#pragma unroll
    for (int ks = 0; ks < 8; ++ks){
        v16bf bw = frag_ld(Woutb, C_DIM, cc * 16 + mA, ks * 32, lane);
        v16bf A0 = frag_ld(accb, C_DIM, p0, ks * 32, lane);
        a0 = wmma_bf16(A0, bw, a0);
        v16bf A1 = frag_ld(accb + (size_t)TPA * C_DIM, C_DIM, p1, ks * 32, lane);
        a1 = wmma_bf16(A1, bw, a1);
    }
    int e = cc * 16 + (lane & 15);
    float ob = outb[e];
#pragma unroll
    for (int vv = 0; vv < 8; ++vv){
        int t = tt * 16 + vv + half * 8;
        int q0 = PAD_W + t;
        int q1 = PAD_W + ((t + SHIFT_W) & (T_SEQ - 1));
        float n0 = win_count(q0), n1 = win_count(q1);
        float y = 0.5f * ((a0[vv] + n0 * ob) / (n0 + 1e-6f) +
                          (a1[vv] + n1 * ob) / (n1 + 1e-6f));
        ycomb[(size_t)t * C_DIM + e] = y;
        ycombb[(size_t)t * C_DIM + e] = f2bfu(y);
    }
}

/* ---------------- channel projection + column-sum for SE mean ---------------- */
__global__ __launch_bounds__(128) void k_proj(const u16t* __restrict__ ycombb,
        const u16t* __restrict__ Wprojb, const float* __restrict__ projb,
        float* __restrict__ yproj, float* __restrict__ colsum){
    int lane = threadIdx.x & 31, wv = threadIdx.x >> 5, half = lane >> 4;
    int tile = blockIdx.x * 4 + wv;
    int tt = tile >> 4, cc = tile & 15;
    int mA = lane & 15;
    v8f a = vzero8();
#pragma unroll
    for (int ks = 0; ks < 8; ++ks){
        v16bf A = frag_ld(ycombb, C_DIM, tt * 16 + mA, ks * 32, lane);
        v16bf bw = frag_ld(Wprojb, C_DIM, cc * 16 + mA, ks * 32, lane);
        a = wmma_bf16(A, bw, a);
    }
    int e = cc * 16 + (lane & 15);
    float pb = projb[e];
#pragma unroll
    for (int vv = 0; vv < 8; ++vv){
        int t = tt * 16 + vv + half * 8;
        float val = a[vv] + pb;
        yproj[(size_t)t * C_DIM + e] = val;
        atomicAdd(&colsum[e], val);
    }
}

/* ---------------- SE gate (256 -> 16 -> 256) ---------------- */
__global__ __launch_bounds__(256) void k_se(const float* __restrict__ colsum,
        const float* __restrict__ w1, const float* __restrict__ w2, float* __restrict__ sig){
    __shared__ float s[256];
    __shared__ float hid[16];
    int tid = threadIdx.x;
    s[tid] = colsum[tid] * (1.0f / (float)T_SEQ);
    __syncthreads();
    if (tid < 16){
        float a = 0.0f;
        for (int e = 0; e < 256; ++e) a += s[e] * w1[tid * 256 + e];
        hid[tid] = fmaxf(a, 0.0f);
    }
    __syncthreads();
    float a = 0.0f;
#pragma unroll
    for (int r = 0; r < 16; ++r) a += hid[r] * w2[tid * 16 + r];
    sig[tid] = 1.0f / (1.0f + __expf(-a));
}

/* ---------------- residual + gate ---------------- */
__global__ void k_final(const float* __restrict__ x, const float* __restrict__ yproj,
                        const float* __restrict__ sig, float* __restrict__ out, int n){
    for (int i = blockIdx.x * blockDim.x + threadIdx.x; i < n; i += gridDim.x * blockDim.x)
        out[i] = x[i] + yproj[i] * sig[i & (C_DIM - 1)];
}

extern "C" void kernel_launch(void* const* d_in, const int* in_sizes, int n_in,
                              void* d_out, int out_size, void* d_ws, size_t ws_size,
                              hipStream_t stream){
    const float* x        = (const float*)d_in[0];
    const float* ln_g     = (const float*)d_in[1];
    const float* ln_b     = (const float*)d_in[2];
    const float* in_projw = (const float*)d_in[3];
    const float* in_projb = (const float*)d_in[4];
    const float* out_w    = (const float*)d_in[5];
    const float* out_b    = (const float*)d_in[6];
    const float* proj_w   = (const float*)d_in[7];
    const float* proj_b   = (const float*)d_in[8];
    const float* se_w1    = (const float*)d_in[9];
    const float* se_w2    = (const float*)d_in[10];
    float* out = (float*)d_out;

    char* ws = (char*)d_ws;
    size_t off = 0;
    auto alloc = [&](size_t bytes) -> void* {
        void* p = (void*)(ws + off);
        off += (bytes + 255) & ~(size_t)255;
        return p;
    };
    u16t*  ylnb   = (u16t*) alloc((size_t)T_SEQ * C_DIM * 2);
    u16t*  Wqkvb  = (u16t*) alloc((size_t)E3 * C_DIM * 2);
    u16t*  Woutb  = (u16t*) alloc((size_t)C_DIM * C_DIM * 2);
    u16t*  Wprojb = (u16t*) alloc((size_t)C_DIM * C_DIM * 2);
    u16t*  QKp    = (u16t*) alloc((size_t)2 * TPA * 512 * 2);
    u16t*  VpT    = (u16t*) alloc((size_t)2 * C_DIM * TPA * 2);
    float* ubuf   = (float*)alloc((size_t)2 * TPA * N_HEADS * 4);
    float* wbuf   = (float*)alloc((size_t)2 * TPA * N_HEADS * 4);
    float* c0buf  = (float*)alloc(N_HEADS * 4);
    float* accatt = (float*)alloc((size_t)2 * TPA * C_DIM * 4);
    u16t*  accb   = (u16t*) alloc((size_t)2 * TPA * C_DIM * 2);
    float* ycomb  = (float*)alloc((size_t)T_SEQ * C_DIM * 4);
    u16t*  ycombb = (u16t*) alloc((size_t)T_SEQ * C_DIM * 2);
    float* yproj  = (float*)alloc((size_t)T_SEQ * C_DIM * 4);
    float* colsum = (float*)alloc(C_DIM * 4);
    float* sig    = (float*)alloc(C_DIM * 4);
    (void)in_sizes; (void)n_in; (void)out_size; (void)ws_size; (void)ycomb;

    int nacc = 2 * TPA * C_DIM;
    k_zero<<<(nacc + 255) / 256, 256, 0, stream>>>(accatt, nacc);
    k_zero<<<1, 256, 0, stream>>>(colsum, C_DIM);

    k_cvt_f2b<<<(E3 * C_DIM + 255) / 256, 256, 0, stream>>>(in_projw, Wqkvb, E3 * C_DIM);
    k_cvt_f2b<<<(C_DIM * C_DIM + 255) / 256, 256, 0, stream>>>(out_w, Woutb, C_DIM * C_DIM);
    k_cvt_f2b<<<(C_DIM * C_DIM + 255) / 256, 256, 0, stream>>>(proj_w, Wprojb, C_DIM * C_DIM);

    k_layernorm<<<T_SEQ, 256, 0, stream>>>(x, ln_g, ln_b, ylnb);

    k_qkv_proj<<<(2 * 72 * 48) / 8, 256, 0, stream>>>(ylnb, Wqkvb, QKp, VpT);

    k_biasdots<<<2 * TPA, 256, 0, stream>>>(QKp, in_projb, ubuf, wbuf);
    k_c0<<<1, 256, 0, stream>>>(in_projb, c0buf);

    k_attn<<<2 * L_WIN * N_HEADS, 256, 0, stream>>>(QKp, VpT, ubuf, wbuf, c0buf, in_projb, accatt);

    k_cvt_f2b<<<(nacc + 255) / 256, 256, 0, stream>>>(accatt, accb, nacc);

    k_outproj<<<(64 * 16) / 4, 128, 0, stream>>>(accb, Woutb, out_b, ycomb, ycombb);
    k_proj<<<(64 * 16) / 4, 128, 0, stream>>>(ycombb, Wprojb, proj_b, yproj, colsum);
    k_se<<<1, 256, 0, stream>>>(colsum, se_w1, se_w2, sig);
    k_final<<<(T_SEQ * C_DIM + 255) / 256, 256, 0, stream>>>(x, yproj, sig, out, T_SEQ * C_DIM);
}
